// CentroidLoss_62947040690217
// MI455X (gfx1250) — matrix operand order repaired and checked by reading, not compiled, not run on hardware
//
#include <hip/hip_runtime.h>
#include <hip/hip_bf16.h>

// Problem constants (match reference)
#define BN 16384
#define CN 2048
#define DN 256
#define THR   0.8f
#define M_NEG 0.3f
#define EPS   1e-8f

typedef __attribute__((ext_vector_type(16))) __bf16 v16bf;
typedef __attribute__((ext_vector_type(8)))  __bf16 v8bf;
typedef __attribute__((ext_vector_type(8)))  float  v8f;

// ---------------------------------------------------------------------------
// Kernel A: per row b — find class index (label >= THR), scatter-add preds
// into centroid sums + counts, and write normalized preds as bf16.
// grid = BN blocks, 256 threads (== DN).
// ---------------------------------------------------------------------------
__global__ void __launch_bounds__(256)
classify_scatter(const float* __restrict__ preds,
                 const float* __restrict__ labels,
                 float* __restrict__ cent_sum,   // [CN*DN]
                 int*   __restrict__ cnt,        // [CN]
                 int*   __restrict__ cls,        // [BN]
                 __bf16* __restrict__ pn_bf)     // [BN*DN]
{
    const int b = blockIdx.x;
    const int t = threadIdx.x;

    __shared__ int   s_cls;
    __shared__ float s_red[256];

    if (t == 0) s_cls = -1;
    __syncthreads();

    const float* lrow = labels + (size_t)b * CN;
    for (int c = t; c < CN; c += 256)
        if (lrow[c] >= THR) s_cls = c;        // one-hot: single writer

    const float p = preds[(size_t)b * DN + t];
    s_red[t] = p * p;
    __syncthreads();                          // also publishes s_cls
    for (int s = 128; s > 0; s >>= 1) {
        if (t < s) s_red[t] += s_red[t + s];
        __syncthreads();
    }
    const float inv = 1.0f / fmaxf(sqrtf(s_red[0]), EPS);
    pn_bf[(size_t)b * DN + t] = (__bf16)(p * inv);

    const int myc = s_cls;
    if (t == 0) cls[b] = myc;
    if (myc >= 0) {
        atomicAdd(&cent_sum[(size_t)myc * DN + t], p);
        if (t == 0) atomicAdd(&cnt[myc], 1);
    }
}

// ---------------------------------------------------------------------------
// Kernel B: finalize centroids: mean -> L2-normalize -> bf16; count existing.
// grid = CN blocks, 256 threads (== DN).
// ---------------------------------------------------------------------------
__global__ void __launch_bounds__(256)
finalize_centroids(const float* __restrict__ cent_sum,
                   const int*   __restrict__ cnt,
                   __bf16* __restrict__ cn_bf,   // [CN*DN]
                   int*    __restrict__ nExist)
{
    const int c = blockIdx.x;
    const int t = threadIdx.x;
    __shared__ float s_red[256];

    const int   n = cnt[c];
    const float v = cent_sum[(size_t)c * DN + t] / (float)max(n, 1);
    s_red[t] = v * v;
    __syncthreads();
    for (int s = 128; s > 0; s >>= 1) {
        if (t < s) s_red[t] += s_red[t + s];
        __syncthreads();
    }
    const float inv = 1.0f / fmaxf(sqrtf(s_red[0]), EPS);
    cn_bf[(size_t)c * DN + t] = (__bf16)(v * inv);
    if (t == 0 && n > 0) atomicAdd(nExist, 1);
}

// ---------------------------------------------------------------------------
// Kernel C: fused WMMA GEMM  cos = pn @ cn^T  with branch-free hinge/pos
// epilogue. grid = BN/16 blocks (one 16-row stripe each), 128 threads = 4
// waves. Wave w sweeps column tiles ct ≡ w (mod 4), TWO tiles per iteration
// (two independent accumulator chains for WMMA/load overlap).
// ---------------------------------------------------------------------------
__global__ void __launch_bounds__(128)
cos_hinge_gemm(const __bf16* __restrict__ pn_bf,
               const __bf16* __restrict__ cn_bf,
               const int*    __restrict__ cls,
               const int*    __restrict__ cnt,
               float* __restrict__ posArr,   // [BN] atomic (single diagonal hit)
               float* __restrict__ negArr)   // [BN] atomic hinge sums
{
    const int stripe  = blockIdx.x;
    const int wave    = threadIdx.x >> 5;
    const int lane    = threadIdx.x & 31;
    const int hi      = lane >> 4;     // 0: lanes 0-15, 1: lanes 16-31
    const int ln      = lane & 15;
    const int rowBase = stripe * 16;

    // cls for the 8 rows this lane's C-fragment elements map to
    int clsLocal[8];
#pragma unroll
    for (int v = 0; v < 8; ++v)
        clsLocal[v] = cls[rowBase + hi * 8 + v];

    float hingeAcc[8] = {0.f, 0.f, 0.f, 0.f, 0.f, 0.f, 0.f, 0.f};
    float posAcc[8]   = {0.f, 0.f, 0.f, 0.f, 0.f, 0.f, 0.f, 0.f};

    // A fragments: lane (either half) owns row rowBase+ln of pn. Load the
    // whole K=256 stripe once into registers; reused across all column tiles.
    const __bf16* aBase = pn_bf + (size_t)(rowBase + ln) * DN;
    v16bf aFrag[8];
#pragma unroll
    for (int kk = 0; kk < 8; ++kk) {
        const int k0 = kk * 32;
        v8bf alo = *(const v8bf*)(aBase + k0 + hi * 8);
        v8bf ahi = *(const v8bf*)(aBase + k0 + 16 + hi * 8);
        aFrag[kk] = __builtin_shufflevector(alo, ahi,
                      0, 1, 2, 3, 4, 5, 6, 7, 8, 9, 10, 11, 12, 13, 14, 15);
    }

    for (int ct = wave; ct < CN / 16; ct += 8) {
        const int ct2      = ct + 4;
        const int colBase0 = ct * 16;
        const int colBase1 = ct2 * 16;
        const __bf16* bBase0 = cn_bf + (size_t)(colBase0 + ln) * DN + hi * 8;
        const __bf16* bBase1 = cn_bf + (size_t)(colBase1 + ln) * DN + hi * 8;

        v8f c0 = {};
        v8f c1 = {};
#pragma unroll
        for (int kk = 0; kk < 8; ++kk) {
            const int k0 = kk * 32;
            v16bf b0 = __builtin_shufflevector(*(const v8bf*)(bBase0 + k0), *(const v8bf*)(bBase0 + k0 + 16),
                      0, 1, 2, 3, 4, 5, 6, 7, 8, 9, 10, 11, 12, 13, 14, 15);
            v16bf b1 = __builtin_shufflevector(*(const v8bf*)(bBase1 + k0), *(const v8bf*)(bBase1 + k0 + 16),
                      0, 1, 2, 3, 4, 5, 6, 7, 8, 9, 10, 11, 12, 13, 14, 15);
            c0 = __builtin_amdgcn_wmma_f32_16x16x32_bf16(
                     false, aFrag[kk], false, b0, (short)0, c0, false, false);
            c1 = __builtin_amdgcn_wmma_f32_16x16x32_bf16(
                     false, aFrag[kk], false, b1, (short)0, c1, false, false);
        }

        const int   ca0 = colBase0 + ln;
        const int   ca1 = colBase1 + ln;
        const float ex0 = (cnt[ca0] > 0) ? 1.0f : 0.0f;
        const float ex1 = (cnt[ca1] > 0) ? 1.0f : 0.0f;
#pragma unroll
        for (int v = 0; v < 8; ++v) {
            const float cv0 = c0[v];
            const float cv1 = c1[v];
            const bool  d0  = (ca0 == clsLocal[v]);   // diagonal?
            const bool  d1  = (ca1 == clsLocal[v]);
            posAcc[v]  += d0 ? (1.0f - cv0) : 0.0f;
            posAcc[v]  += d1 ? (1.0f - cv1) : 0.0f;
            hingeAcc[v] += (d0 ? 0.0f : ex0) * fmaxf(cv0 - (1.0f - M_NEG), 0.0f);
            hingeAcc[v] += (d1 ? 0.0f : ex1) * fmaxf(cv1 - (1.0f - M_NEG), 0.0f);
        }
    }

    // Reduce across the 16 lanes sharing each row (xor masks < 16 keep halves)
#pragma unroll
    for (int v = 0; v < 8; ++v) {
        float hx = hingeAcc[v];
        float px = posAcc[v];
#pragma unroll
        for (int off = 1; off < 16; off <<= 1) {
            hx += __shfl_xor(hx, off, 32);
            px += __shfl_xor(px, off, 32);
        }
        if (ln == 0) {
            const int r = rowBase + hi * 8 + v;
            atomicAdd(&negArr[r], hx);
            atomicAdd(&posArr[r], px);   // exactly one nonzero contribution total
        }
    }
}

// ---------------------------------------------------------------------------
// Kernel D: scalar = mean(pos) + mean(negSum / max(nExist-1, 1)).
// Single block of 256 threads.
// ---------------------------------------------------------------------------
__global__ void __launch_bounds__(256)
final_reduce(const float* __restrict__ posArr,
             const float* __restrict__ negArr,
             const int*   __restrict__ nExist,
             float* __restrict__ out)
{
    __shared__ float s[256];
    const int t = threadIdx.x;
    const float invAdds = 1.0f / (float)max(*nExist - 1, 1);

    float acc = 0.f;
    for (int b = t; b < BN; b += 256)
        acc += posArr[b] + negArr[b] * invAdds;

    s[t] = acc;
    __syncthreads();
    for (int sft = 128; sft > 0; sft >>= 1) {
        if (t < sft) s[t] += s[t + sft];
        __syncthreads();
    }
    if (t == 0) out[0] = s[0] / (float)BN;
}

// ---------------------------------------------------------------------------
// Host launcher
// ---------------------------------------------------------------------------
extern "C" void kernel_launch(void* const* d_in, const int* in_sizes, int n_in,
                              void* d_out, int out_size, void* d_ws, size_t ws_size,
                              hipStream_t stream)
{
    (void)in_sizes; (void)n_in; (void)out_size; (void)ws_size;

    const float* preds  = (const float*)d_in[0];   // [BN, DN]
    const float* labels = (const float*)d_in[1];   // [BN, CN]
    float* out = (float*)d_out;                    // scalar

    char* ws = (char*)d_ws;
    const size_t OFF_CENT = 0;                              // CN*DN f32
    const size_t OFF_CNT  = OFF_CENT + (size_t)CN * DN * 4; // CN i32
    const size_t OFF_NEX  = OFF_CNT  + (size_t)CN * 4;      // 1 i32 (padded)
    const size_t OFF_NEG  = OFF_NEX  + 256;                 // BN f32
    const size_t OFF_POS  = OFF_NEG  + (size_t)BN * 4;      // BN f32
    const size_t OFF_CLS  = OFF_POS  + (size_t)BN * 4;      // BN i32
    const size_t OFF_CNBF = OFF_CLS  + (size_t)BN * 4;      // CN*DN bf16
    const size_t OFF_PNBF = OFF_CNBF + (size_t)CN * DN * 2; // BN*DN bf16

    float*  cent_sum = (float*)(ws + OFF_CENT);
    int*    cnt      = (int*)  (ws + OFF_CNT);
    int*    nExist   = (int*)  (ws + OFF_NEX);
    float*  negArr   = (float*)(ws + OFF_NEG);
    float*  posArr   = (float*)(ws + OFF_POS);
    int*    cls      = (int*)  (ws + OFF_CLS);
    __bf16* cn_bf    = (__bf16*)(ws + OFF_CNBF);
    __bf16* pn_bf    = (__bf16*)(ws + OFF_PNBF);

    // Zero: [cent_sum | cnt | nExist] and [negArr | posArr] (graph-capturable)
    hipMemsetAsync(ws + OFF_CENT, 0, OFF_NEG - OFF_CENT, stream);
    hipMemsetAsync(ws + OFF_NEG,  0, (size_t)BN * 8, stream);

    classify_scatter<<<BN, 256, 0, stream>>>(preds, labels, cent_sum, cnt, cls, pn_bf);
    finalize_centroids<<<CN, 256, 0, stream>>>(cent_sum, cnt, cn_bf, nExist);
    cos_hinge_gemm<<<BN / 16, 128, 0, stream>>>(pn_bf, cn_bf, cls, cnt, posArr, negArr);
    final_reduce<<<1, 256, 0, stream>>>(posArr, negArr, nExist, out);
}
